// SelfAttention_12541304504837
// MI455X (gfx1250) — matrix operand order repaired and checked by reading, not compiled, not run on hardware
//
#include <hip/hip_runtime.h>
#include <hip/hip_bf16.h>
#include <stdint.h>

// Problem constants (match reference)
#define B_    4
#define C_    256
#define CQK_  32
#define N_    4096   // 64*64

typedef __attribute__((ext_vector_type(16))) __bf16   v16bf;
typedef __attribute__((ext_vector_type(8)))  float    v8f;
typedef __attribute__((ext_vector_type(4)))  uint32_t v4u;

union ABf16 { v16bf v; uint32_t u[8]; v4u q[2]; };

__device__ __forceinline__ uint16_t f2bf(float f) {
  uint32_t u = __float_as_uint(f);
  u += 0x7FFFu + ((u >> 16) & 1u);      // round-to-nearest-even
  return (uint16_t)(u >> 16);
}

// v_permlane16_b32 xor-shuffle within 16-lane rows (VALU; no DS waits).
__device__ __forceinline__ float pl16(float v, int s1, int s2) {
  int r = __builtin_amdgcn_permlane16(__float_as_int(v), __float_as_int(v),
                                      s1, s2, false, false);
  return __int_as_float(r);
}

// ---------------------------------------------------------------------------
// Kernel 1: fused Q/K/V 1x1-conv projections, f32 in -> bf16 out.
//   Q, K stored [B][N][32]  (A-tile / K^T B-tile friendly)
//   V    stored [B][C][N]   (PV B-tile friendly)
// ---------------------------------------------------------------------------
#define PIX_ 8
__global__ __launch_bounds__(256)
void qkv_proj_kernel(const float* __restrict__ x,
                     const float* __restrict__ wq, const float* __restrict__ bq,
                     const float* __restrict__ wk, const float* __restrict__ bk,
                     const float* __restrict__ wv, const float* __restrict__ bv,
                     uint16_t* __restrict__ Qb, uint16_t* __restrict__ Kb,
                     uint16_t* __restrict__ Vb) {
  __shared__ float xs[C_][PIX_ + 1];
  const int t   = threadIdx.x;                 // channel 0..255
  const int blk = blockIdx.x;                  // 0 .. B*(N/PIX_)-1
  const int b   = blk / (N_ / PIX_);
  const int n0  = (blk % (N_ / PIX_)) * PIX_;

  const float* xrow = x + ((size_t)(b * C_ + t)) * N_ + n0;
#pragma unroll
  for (int p = 0; p < PIX_; ++p) xs[t][p] = xrow[p];
  __syncthreads();

  float av[PIX_];
#pragma unroll
  for (int p = 0; p < PIX_; ++p) av[p] = bv[t];
  const float* wvr = wv + (size_t)t * C_;
  for (int c = 0; c < C_; ++c) {
    float w = wvr[c];
#pragma unroll
    for (int p = 0; p < PIX_; ++p) av[p] = fmaf(w, xs[c][p], av[p]);
  }
  uint16_t* vdst = Vb + ((size_t)(b * C_ + t)) * N_ + n0;
#pragma unroll
  for (int p = 0; p < PIX_; ++p) vdst[p] = f2bf(av[p]);

  if (t < CQK_) {
    float aq[PIX_], ak[PIX_];
#pragma unroll
    for (int p = 0; p < PIX_; ++p) { aq[p] = bq[t]; ak[p] = bk[t]; }
    const float* wqr = wq + (size_t)t * C_;
    const float* wkr = wk + (size_t)t * C_;
    for (int c = 0; c < C_; ++c) {
      float wq_ = wqr[c], wk_ = wkr[c];
#pragma unroll
      for (int p = 0; p < PIX_; ++p) {
        aq[p] = fmaf(wq_, xs[c][p], aq[p]);
        ak[p] = fmaf(wk_, xs[c][p], ak[p]);
      }
    }
#pragma unroll
    for (int p = 0; p < PIX_; ++p) {
      Qb[((size_t)(b * N_ + n0 + p)) * CQK_ + t] = f2bf(aq[p]);
      Kb[((size_t)(b * N_ + n0 + p)) * CQK_ + t] = f2bf(ak[p]);
    }
  }
}

// ---------------------------------------------------------------------------
// Kernel 2: cooperative flash attention. 4 waves/block share a 16-row m-tile.
// Superstep = 128 columns: wave w softmaxes only its 32-column sub-block and
// publishes P (bf16) / row-max / row-sum through LDS; every wave consumes all
// 128 columns of P against its own 4 c-tiles of the 16x256 O accumulator.
// ---------------------------------------------------------------------------
#define SS_ 128                                 // columns per superstep
__global__ __launch_bounds__(128)
void attn_kernel(const uint16_t* __restrict__ Qb, const uint16_t* __restrict__ Kb,
                 const uint16_t* __restrict__ Vb, const float* __restrict__ x,
                 const float* __restrict__ gamma_p, float* __restrict__ out) {
  __shared__ uint16_t Pbuf[16][SS_];            // shared P tile (4 KiB)
  __shared__ float    lmax[4][16];              // per-wave local row max
  __shared__ float    rsum[4][16];              // per-wave local row sum

  const int w    = threadIdx.x >> 5;            // wave id 0..3
  const int lane = threadIdx.x & 31;
  const int half = lane >> 4;
  const int lc   = lane & 15;
  const int blk  = blockIdx.x;
  const int b    = blk / (N_ / 16);
  const int m0   = (blk % (N_ / 16)) * 16;
  const int ctb  = w * 4;                       // this wave's first c-tile

  // Q A-tile (16 rows x K=32)
  ABf16 qa;
  {
    const uint32_t qoff = (uint32_t)(b * N_ + m0 + lc) * CQK_ + (half << 3);
    qa.q[0] = *(const v4u*)(Qb + qoff);
    qa.q[1] = *(const v4u*)(Qb + qoff + 16);
  }

  ABf16 onesb;                                  // all-ones B matrix (row-sum)
#pragma unroll
  for (int v = 0; v < 8; ++v) onesb.u[v] = 0x3F803F80u;

  v8f zero = {0.f, 0.f, 0.f, 0.f, 0.f, 0.f, 0.f, 0.f};
  v8f o[4];
#pragma unroll
  for (int cc = 0; cc < 4; ++cc) o[cc] = zero;
  float mrun[8], lsum[8];
#pragma unroll
  for (int i = 0; i < 8; ++i) { mrun[i] = -INFINITY; lsum[i] = 0.f; }

  for (int j = 0; j < N_ / SS_; ++j) {
    const int n0 = j * SS_;
    const int ns = n0 + w * 32;                 // this wave's sub-block

    // prefetch next superstep's K sub-block (global_prefetch_b8)
    if (j + 1 < N_ / SS_) {
      __builtin_prefetch(Kb + (uint32_t)(b * N_ + ns + SS_ + lc) * CQK_, 0, 1);
    }

    // ---- Phase 1: S for own 32 columns + local row max --------------------
    ABf16 kb0, kb1;
    {
      const uint32_t koff = (uint32_t)(b * N_ + ns + lc) * CQK_ + (half << 3);
      kb0.q[0] = *(const v4u*)(Kb + koff);
      kb0.q[1] = *(const v4u*)(Kb + koff + 16);
      kb1.q[0] = *(const v4u*)(Kb + koff + 16 * CQK_);
      kb1.q[1] = *(const v4u*)(Kb + koff + 16 * CQK_ + 16);
    }
    v8f s0 = __builtin_amdgcn_wmma_f32_16x16x32_bf16(false, qa.v, false, kb0.v,
                                                     (short)0, zero, false, false);
    v8f s1 = __builtin_amdgcn_wmma_f32_16x16x32_bf16(false, qa.v, false, kb1.v,
                                                     (short)0, zero, false, false);
    float lm[8];
#pragma unroll
    for (int i = 0; i < 8; ++i) {
      float c = fmaxf(s0[i], s1[i]);
      c = fmaxf(c, pl16(c, (int)0x67452301, (int)0xEFCDAB89));  // xor 1
      c = fmaxf(c, pl16(c, (int)0x54761032, (int)0xDCFE98BA));  // xor 2
      c = fmaxf(c, pl16(c, (int)0x32107654, (int)0xBA98FEDC));  // xor 4
      c = fmaxf(c, pl16(c, (int)0xFEDCBA98, (int)0x76543210));  // xor 8
      lm[i] = c;
    }
    if (lc == 0) {
#pragma unroll
      for (int i = 0; i < 8; ++i) lmax[w][(half << 3) + i] = lm[i];
    }
    __syncthreads();

    // ---- Phase 2: combined max, exp own block, publish P + row-sum --------
    float cm[8];
#pragma unroll
    for (int i = 0; i < 8; ++i) cm[i] = -INFINITY;
#pragma unroll
    for (int ww = 0; ww < 4; ++ww) {
      float4 a = *(const float4*)&lmax[ww][half << 3];
      float4 c = *(const float4*)&lmax[ww][(half << 3) + 4];
      cm[0] = fmaxf(cm[0], a.x); cm[1] = fmaxf(cm[1], a.y);
      cm[2] = fmaxf(cm[2], a.z); cm[3] = fmaxf(cm[3], a.w);
      cm[4] = fmaxf(cm[4], c.x); cm[5] = fmaxf(cm[5], c.y);
      cm[6] = fmaxf(cm[6], c.z); cm[7] = fmaxf(cm[7], c.w);
    }
    float scale[8];
#pragma unroll
    for (int i = 0; i < 8; ++i) {
      float mn = fmaxf(mrun[i], cm[i]);
      scale[i] = __expf(mrun[i] - mn);
      mrun[i]  = mn;
      s0[i] = __expf(s0[i] - mn);
      s1[i] = __expf(s1[i] - mn);
    }
#pragma unroll
    for (int i = 0; i < 8; ++i) {
      int r = (half << 3) + i;
      Pbuf[r][w * 32 + lc]      = f2bf(s0[i]);
      Pbuf[r][w * 32 + 16 + lc] = f2bf(s1[i]);
    }
    // row-sum of own sub-block via ones-matrix WMMA
    ABf16 paw;
    {
      const uint16_t* prow = &Pbuf[lc][w * 32 + (half << 3)];
      paw.q[0] = *(const v4u*)(prow);
      paw.q[1] = *(const v4u*)(prow + 16);
    }
    v8f rsv = __builtin_amdgcn_wmma_f32_16x16x32_bf16(false, paw.v, false, onesb.v,
                                                      (short)0, zero, false, false);
    if (lc == 0) {
#pragma unroll
      for (int i = 0; i < 8; ++i) rsum[w][(half << 3) + i] = rsv[i];
    }
    __syncthreads();

    // ---- Phase 3: fold row-sums, rescale O, P*V over all 128 columns ------
    float rs[8];
#pragma unroll
    for (int i = 0; i < 8; ++i) rs[i] = 0.f;
#pragma unroll
    for (int ww = 0; ww < 4; ++ww) {
      float4 a = *(const float4*)&rsum[ww][half << 3];
      float4 c = *(const float4*)&rsum[ww][(half << 3) + 4];
      rs[0] += a.x; rs[1] += a.y; rs[2] += a.z; rs[3] += a.w;
      rs[4] += c.x; rs[5] += c.y; rs[6] += c.z; rs[7] += c.w;
    }
#pragma unroll
    for (int i = 0; i < 8; ++i) lsum[i] = lsum[i] * scale[i] + rs[i];
#pragma unroll
    for (int cc = 0; cc < 4; ++cc) {
#pragma unroll
      for (int i = 0; i < 8; ++i) o[cc][i] *= scale[i];
    }

    ABf16 pa[4];
#pragma unroll
    for (int sb = 0; sb < 4; ++sb) {
      const uint16_t* prow = &Pbuf[lc][sb * 32 + (half << 3)];
      pa[sb].q[0] = *(const v4u*)(prow);
      pa[sb].q[1] = *(const v4u*)(prow + 16);
    }
#pragma unroll
    for (int cc = 0; cc < 4; ++cc) {
      const uint32_t vbase = (uint32_t)(b * C_ + (ctb + cc) * 16 + lc) * N_ + n0;
#pragma unroll
      for (int sb = 0; sb < 4; ++sb) {
        ABf16 vbt;
        const uint32_t voff = vbase + sb * 32 + (half << 3);
        vbt.q[0] = *(const v4u*)(Vb + voff);
        vbt.q[1] = *(const v4u*)(Vb + voff + 16);
        o[cc] = __builtin_amdgcn_wmma_f32_16x16x32_bf16(false, pa[sb].v, false, vbt.v,
                                                        (short)0, o[cc], false, false);
      }
    }
    // (no third barrier needed: Pbuf/rsum rewrites happen in the next
    //  superstep's Phase 2, which all waves reach only after Phase-1's
    //  __syncthreads, i.e. after every wave finished this Phase 3.)
  }

  // ---- finalize: out[b,c,m] = gamma * O[m,c]/l[m] + x[b,c,m] --------------
  const float g = gamma_p[0];
  float rinv[8];
#pragma unroll
  for (int i = 0; i < 8; ++i) rinv[i] = 1.0f / lsum[i];

#pragma unroll
  for (int cc = 0; cc < 4; ++cc) {
    const int c = (ctb + cc) * 16 + lc;
    const uint32_t base = (uint32_t)(b * C_ + c) * N_ + m0 + (half << 3);
#pragma unroll
    for (int i = 0; i < 8; ++i) {
      out[base + i] = g * (o[cc][i] * rinv[i]) + x[base + i];
    }
  }
}

// ---------------------------------------------------------------------------
// Host launcher. Workspace (bf16): Qb 1 MiB | Kb 1 MiB | Vb 8 MiB = 10 MiB.
// ---------------------------------------------------------------------------
extern "C" void kernel_launch(void* const* d_in, const int* in_sizes, int n_in,
                              void* d_out, int out_size, void* d_ws, size_t ws_size,
                              hipStream_t stream) {
  const float* x     = (const float*)d_in[0];
  const float* wq    = (const float*)d_in[1];
  const float* bq    = (const float*)d_in[2];
  const float* wk    = (const float*)d_in[3];
  const float* bk    = (const float*)d_in[4];
  const float* wv    = (const float*)d_in[5];
  const float* bv    = (const float*)d_in[6];
  const float* gamma = (const float*)d_in[7];
  float* out = (float*)d_out;

  uint16_t* Qb = (uint16_t*)d_ws;
  uint16_t* Kb = Qb + (size_t)B_ * N_ * CQK_;
  uint16_t* Vb = Kb + (size_t)B_ * N_ * CQK_;

  qkv_proj_kernel<<<B_ * (N_ / PIX_), 256, 0, stream>>>(
      x, wq, bq, wk, bk, wv, bv, Qb, Kb, Vb);

  attn_kernel<<<B_ * (N_ / 16), 128, 0, stream>>>(
      Qb, Kb, Vb, x, gamma, out);
}
